// RetrievalAugmentor_35536559407763
// MI455X (gfx1250) — matrix operand order repaired and checked by reading, not compiled, not run on hardware
//
#include <hip/hip_runtime.h>
#include <hip/hip_bf16.h>
#include <math.h>

#define BQ 256
#define NN 100000
#define ED 1280
#define DD 512
#define KK 8
#define NT 64            // index rows per workgroup in the sims GEMM

typedef __attribute__((ext_vector_type(16))) __bf16 v16bf;
typedef __attribute__((ext_vector_type(8)))  float  v8f;

struct U4 { unsigned int x, y, z, w; };           // trivial 16-byte chunk
struct U2 { unsigned int a, b; };                 // trivial 8-byte chunk
struct F4 { float x, y, z, w; };                  // trivial 16-byte f32 chunk
union Frag16 { v16bf v; U4 q[2]; };

__device__ __forceinline__ unsigned short f2bf(float f) {
  unsigned int u = __float_as_uint(f);
  unsigned int r = (u + 0x7FFFu + ((u >> 16) & 1u)) >> 16;   // RNE
  return (unsigned short)r;
}

__device__ __forceinline__ U2 pack4bf(F4 f) {
  U2 u;
  u.a = (unsigned)f2bf(f.x) | ((unsigned)f2bf(f.y) << 16);
  u.b = (unsigned)f2bf(f.z) | ((unsigned)f2bf(f.w) << 16);
  return u;
}

// ---------------- prep: l2-normalize queries -> bf16 ----------------
__global__ void ra_prep_q(const float* __restrict__ q, unsigned short* __restrict__ qbf) {
  __shared__ float red[256];
  int b = blockIdx.x, t = threadIdx.x;
  const float* row = q + (size_t)b * ED;
  float ss = 0.f;
  for (int i = t; i < ED; i += 256) { float v = row[i]; ss += v * v; }
  red[t] = ss; __syncthreads();
  for (int s = 128; s > 0; s >>= 1) { if (t < s) red[t] += red[t + s]; __syncthreads(); }
  float inv = 1.f / fmaxf(sqrtf(red[0]), 1e-12f);
  for (int i = t; i < ED; i += 256) qbf[(size_t)b * ED + i] = f2bf(row[i] * inv);
}

// ---------------- prep: transpose proj_W [E,D] -> bf16 [D,E] ----------------
__global__ void ra_prep_w(const float* __restrict__ W, unsigned short* __restrict__ Wt) {
  int idx = blockIdx.x * 256 + threadIdx.x;
  if (idx >= ED * DD) return;
  int k = idx / DD, n = idx % DD;
  Wt[(size_t)n * ED + k] = f2bf(W[idx]);
}

// ---------------- sims = Q @ index^T  (bf16 WMMA, f32 acc) ----------------
// grid = ceil(N/64); each workgroup owns 64 index rows x all 256 queries.
// 160 KB LDS tile (CDNA5: up to 320 KB per workgroup) -> Q re-read from L2
// drops to ~1 GB total; 8 WMMAs per 4 global A-loads per wave K-step.
__global__ void __launch_bounds__(256) ra_sims(const unsigned short* __restrict__ qbf,
                                               const float* __restrict__ index_emb,
                                               float* __restrict__ sims) {
  __shared__ unsigned short tile[NT * ED];   // 160KB: 64 index rows, bf16
  int t = threadIdx.x;
  int n0 = blockIdx.x * NT;
  // vectorized fill: float4 -> packed bf16 x4 -> 8-byte LDS store
  const int CH = (NT * ED) / 4;
  for (int c = t; c < CH; c += 256) {
    int r  = c / (ED / 4);
    int k4 = (c % (ED / 4)) * 4;
    int row = n0 + r;
    F4 f;
    if (row < NN) f = *(const F4*)(index_emb + (size_t)row * ED + k4);
    else { f.x = 0.f; f.y = 0.f; f.z = 0.f; f.w = 0.f; }
    *(U2*)(tile + (size_t)c * 4) = pack4bf(f);
  }
  __syncthreads();
  int wave = t >> 5, lane = t & 31, half = lane >> 4, l16 = lane & 15;
  v8f c0[4] = {{}, {}, {}, {}};
  v8f c1[4] = {{}, {}, {}, {}};
  const unsigned short* q0 = qbf + (size_t)((wave * 2) * 16 + l16) * ED;
  const unsigned short* q1 = qbf + (size_t)((wave * 2 + 1) * 16 + l16) * ED;
  for (int kb = 0; kb < ED; kb += 32) {
    Frag16 a0, a1;
    int ao = kb + (half ? 8 : 0);                 // 16-bit A layout
    a0.q[0] = *(const U4*)(q0 + ao);
    a0.q[1] = *(const U4*)(q0 + ao + 16);
    a1.q[0] = *(const U4*)(q1 + ao);
    a1.q[1] = *(const U4*)(q1 + ao + 16);
    #pragma unroll
    for (int j = 0; j < 4; j++) {
      Frag16 bfr;                                  // B: same 16-bit K layout as A
      const unsigned short* bp = tile + (size_t)(j * 16 + l16) * ED + kb + half * 8;
      bfr.q[0] = *(const U4*)(bp);
      bfr.q[1] = *(const U4*)(bp + 16);
      c0[j] = __builtin_amdgcn_wmma_f32_16x16x32_bf16(false, a0.v, false, bfr.v, (short)0, c0[j], false, false);
      c1[j] = __builtin_amdgcn_wmma_f32_16x16x32_bf16(false, a1.v, false, bfr.v, (short)0, c1[j], false, false);
    }
  }
  #pragma unroll
  for (int j = 0; j < 4; j++) {
    int col = n0 + j * 16 + l16;
    if (col < NN) {
      #pragma unroll
      for (int r = 0; r < 8; r++) {
        int row0 = (wave * 2) * 16 + r + 8 * half;
        int row1 = (wave * 2 + 1) * 16 + r + 8 * half;
        sims[(size_t)row0 * NN + col] = c0[j][r];
        sims[(size_t)row1 * NN + col] = c1[j][r];
      }
    }
  }
}

// ---------------- per-row top-8 with exclusion ----------------
__global__ void ra_topk(const float* __restrict__ sims, const int* __restrict__ excl,
                        int* __restrict__ topk) {
  __shared__ float sv[2048];
  __shared__ int   si[2048];
  __shared__ float rv[256];
  __shared__ int   ri[256];
  int b = blockIdx.x, t = threadIdx.x;
  int ex = excl[b];
  float v[8]; int id[8];
  #pragma unroll
  for (int j = 0; j < 8; j++) { v[j] = -1e30f; id[j] = -1; }
  const float* row = sims + (size_t)b * NN;
  for (int n = t; n < NN; n += 256) {
    if (n == ex) continue;
    float s = row[n];
    if (s > v[7]) {
      int p = 7;
      while (p > 0 && s > v[p - 1]) { v[p] = v[p - 1]; id[p] = id[p - 1]; p--; }
      v[p] = s; id[p] = n;
    }
  }
  #pragma unroll
  for (int j = 0; j < 8; j++) { sv[t * 8 + j] = v[j]; si[t * 8 + j] = id[j]; }
  __syncthreads();
  for (int sel = 0; sel < 8; sel++) {
    float mv = -1e30f; int mp = t * 8;
    #pragma unroll
    for (int j = 0; j < 8; j++) {
      float x = sv[t * 8 + j];
      if (x > mv) { mv = x; mp = t * 8 + j; }
    }
    rv[t] = mv; ri[t] = mp; __syncthreads();
    for (int s = 128; s > 0; s >>= 1) {
      if (t < s && rv[t + s] > rv[t]) { rv[t] = rv[t + s]; ri[t] = ri[t + s]; }
      __syncthreads();
    }
    if (t == 0) { topk[b * 8 + sel] = si[ri[0]]; sv[ri[0]] = -1e31f; }
    __syncthreads();
  }
}

// ---------------- neigh_proj = gather(index_emb)[2048,1280] @ proj_W + b ----------------
__global__ void __launch_bounds__(256) ra_proj(const int* __restrict__ topk,
                                               const float* __restrict__ index_emb,
                                               const unsigned short* __restrict__ Wt,
                                               const float* __restrict__ bias,
                                               float* __restrict__ out) {
  __shared__ unsigned short tile[16 * ED];   // 16 gathered rows as bf16 A-tile
  int t = threadIdx.x;
  int mbase = blockIdx.x * 16;
  const int CH = (16 * ED) / 4;
  for (int c = t; c < CH; c += 256) {
    int r  = c / (ED / 4);
    int k4 = (c % (ED / 4)) * 4;
    int g = topk[mbase + r];
    F4 f = *(const F4*)(index_emb + (size_t)g * ED + k4);
    *(U2*)(tile + (size_t)c * 4) = pack4bf(f);
  }
  __syncthreads();
  int wave = t >> 5, lane = t & 31, half = lane >> 4, l16 = lane & 15;
  v8f c[4] = {{}, {}, {}, {}};
  const unsigned short* ap = tile + (size_t)l16 * ED;
  for (int kb = 0; kb < ED; kb += 32) {
    Frag16 a;
    int ao = kb + (half ? 8 : 0);
    a.q[0] = *(const U4*)(ap + ao);
    a.q[1] = *(const U4*)(ap + ao + 16);
    #pragma unroll
    for (int j = 0; j < 4; j++) {
      int col = (wave * 4 + j) * 16 + l16;
      Frag16 bfr;
      const unsigned short* bp = Wt + (size_t)col * ED + kb + half * 8;
      bfr.q[0] = *(const U4*)(bp);
      bfr.q[1] = *(const U4*)(bp + 16);
      c[j] = __builtin_amdgcn_wmma_f32_16x16x32_bf16(false, a.v, false, bfr.v, (short)0, c[j], false, false);
    }
  }
  #pragma unroll
  for (int j = 0; j < 4; j++) {
    int col = (wave * 4 + j) * 16 + l16;
    float bb = bias[col];
    #pragma unroll
    for (int r = 0; r < 8; r++) {
      int m = mbase + r + 8 * half;
      out[(size_t)m * DD + col] = c[j][r] + bb;
    }
  }
}

// ---------------- per-row attention + prototypes + logits + loss partial ----------------
__global__ void ra_attn(const float* __restrict__ model_embs,
                        const float* __restrict__ proj,   // [B*8, D]
                        const int* __restrict__ topk,
                        const int* __restrict__ labels,
                        const float* __restrict__ clfW,   // [2D, 2]
                        const float* __restrict__ clfB,
                        float* __restrict__ d_logits,     // d_out[0 .. 2B)
                        float* __restrict__ losspart) {
  __shared__ float red[256];
  __shared__ float att[8];
  __shared__ int   lbl[8];
  __shared__ float cnt[2];
  int b = blockIdx.x, t = threadIdx.x;
  int d0 = t * 2;
  float m0 = model_embs[(size_t)b * DD + d0];
  float m1 = model_embs[(size_t)b * DD + d0 + 1];
  const float* P = proj + (size_t)b * 8 * DD;
  float scale = 1.0f / sqrtf((float)DD);
  float sK[8];
  for (int k = 0; k < 8; k++) {
    float part = m0 * P[k * DD + d0] + m1 * P[k * DD + d0 + 1];
    red[t] = part; __syncthreads();
    for (int s = 128; s > 0; s >>= 1) { if (t < s) red[t] += red[t + s]; __syncthreads(); }
    sK[k] = red[0] * scale;
    __syncthreads();
  }
  if (t == 0) {
    float mx = sK[0];
    for (int k = 1; k < 8; k++) mx = fmaxf(mx, sK[k]);
    float se = 0.f, e[8];
    for (int k = 0; k < 8; k++) { e[k] = __expf(sK[k] - mx); se += e[k]; }
    float cp = 0.f, cn = 0.f;
    for (int k = 0; k < 8; k++) {
      att[k] = e[k] / se;
      int L = labels[topk[b * 8 + k]];
      lbl[k] = L;
      cp += (L == 1) ? 1.f : 0.f;
      cn += (L == 0) ? 1.f : 0.f;
    }
    cnt[0] = cp; cnt[1] = cn;
  }
  __syncthreads();
  float z0 = 0.f, z1 = 0.f, p0 = 0.f, p1 = 0.f, q0 = 0.f, q1 = 0.f;
  for (int k = 0; k < 8; k++) {
    float a = att[k];
    float x0 = P[k * DD + d0], x1 = P[k * DD + d0 + 1];
    z0 += a * x0; z1 += a * x1;
    if (lbl[k] == 1) { p0 += x0; p1 += x1; } else { q0 += x0; q1 += x1; }
  }
  float pos0 = p0 / (cnt[0] + 1e-6f), pos1 = p1 / (cnt[0] + 1e-6f);
  float neg0 = q0 / (cnt[1] + 1e-6f), neg1 = q1 / (cnt[1] + 1e-6f);
  float lp = (m0 - pos0) * (m0 - pos0) + (m1 - pos1) * (m1 - pos1)
           - 0.5f * ((m0 - neg0) * (m0 - neg0) + (m1 - neg1) * (m1 - neg1));
  red[t] = lp; __syncthreads();
  for (int s = 128; s > 0; s >>= 1) { if (t < s) red[t] += red[t + s]; __syncthreads(); }
  if (t == 0) losspart[b] = red[0];
  __syncthreads();
  for (int j = 0; j < 2; j++) {
    float part = m0 * clfW[d0 * 2 + j] + m1 * clfW[(d0 + 1) * 2 + j]
               + z0 * clfW[(DD + d0) * 2 + j] + z1 * clfW[(DD + d0 + 1) * 2 + j];
    red[t] = part; __syncthreads();
    for (int s = 128; s > 0; s >>= 1) { if (t < s) red[t] += red[t + s]; __syncthreads(); }
    if (t == 0) d_logits[b * 2 + j] = red[0] + clfB[j];
    __syncthreads();
  }
}

// ---------------- final loss reduce ----------------
__global__ void ra_final(const float* __restrict__ losspart, float* __restrict__ out) {
  __shared__ float red[256];
  int t = threadIdx.x;
  red[t] = losspart[t]; __syncthreads();
  for (int s = 128; s > 0; s >>= 1) { if (t < s) red[t] += red[t + s]; __syncthreads(); }
  if (t == 0) out[BQ * 2] = red[0] / (float)(BQ * DD);
}

extern "C" void kernel_launch(void* const* d_in, const int* in_sizes, int n_in,
                              void* d_out, int out_size, void* d_ws, size_t ws_size,
                              hipStream_t stream) {
  (void)in_sizes; (void)n_in; (void)out_size; (void)ws_size;
  const float* q     = (const float*)d_in[0];
  const float* memb  = (const float*)d_in[1];
  const float* iemb  = (const float*)d_in[2];
  const int*   ilab  = (const int*)d_in[3];
  const int*   excl  = (const int*)d_in[4];
  const float* projW = (const float*)d_in[5];
  const float* projB = (const float*)d_in[6];
  const float* clfW  = (const float*)d_in[7];
  const float* clfB  = (const float*)d_in[8];
  float* out = (float*)d_out;

  char* ws = (char*)d_ws;
  size_t oSims = 0;                                   // 102.4 MB sims
  size_t oQbf  = oSims + (size_t)BQ * NN * 4;         // 640 KB Q bf16
  size_t oWt   = oQbf + (size_t)BQ * ED * 2;          // 1.31 MB W^T bf16
  size_t oTopk = oWt + (size_t)DD * ED * 2;           // 8 KB topk idx
  size_t oProj = oTopk + (size_t)BQ * KK * 4;         // 4 MB neigh_proj
  size_t oLoss = oProj + (size_t)BQ * KK * DD * 4;    // 1 KB loss partials

  float*          sims  = (float*)(ws + oSims);
  unsigned short* qbf   = (unsigned short*)(ws + oQbf);
  unsigned short* Wt    = (unsigned short*)(ws + oWt);
  int*            topk  = (int*)(ws + oTopk);
  float*          nproj = (float*)(ws + oProj);
  float*          lossp = (float*)(ws + oLoss);

  ra_prep_q<<<BQ, 256, 0, stream>>>(q, qbf);
  ra_prep_w<<<(ED * DD + 255) / 256, 256, 0, stream>>>(projW, Wt);
  ra_sims<<<(NN + NT - 1) / NT, 256, 0, stream>>>(qbf, iemb, sims);
  ra_topk<<<BQ, 256, 0, stream>>>(sims, excl, topk);
  ra_proj<<<(BQ * KK) / 16, 256, 0, stream>>>(topk, iemb, Wt, projB, nproj);
  ra_attn<<<BQ, 256, 0, stream>>>(memb, nproj, topk, ilab, clfW, clfB, out, lossp);
  ra_final<<<1, 256, 0, stream>>>(lossp, out);
}
